// VQVAE_4080218931432
// MI455X (gfx1250) — matrix operand order repaired and checked by reading, not compiled, not run on hardware
//
#include <hip/hip_runtime.h>
#include <hip/hip_bf16.h>

// ---------------------------------------------------------------------------
// VQ-VAE forward on MI455X (gfx1250): all GEMMs on v_wmma_f32_16x16x32_bf16,
// GEMM tile staging via GLOBAL_LOAD_ASYNC_TO_LDS_B128 (ASYNCcnt) with
// double-buffered LDS so global latency overlaps the WMMA pipe.
// B=16384, D_IN=4096, D_H=2048, D_Z=256, K=8192, COMMIT=0.25
// ---------------------------------------------------------------------------

#define BATCH 16384
#define DIN   4096
#define DH    2048
#define DZ    256
#define KCB   8192

typedef __attribute__((ext_vector_type(16))) __bf16 v16bf;
typedef __attribute__((ext_vector_type(8)))  __bf16 v8bf;
typedef __attribute__((ext_vector_type(4)))  __bf16 v4bf;
typedef __attribute__((ext_vector_type(8)))  float  v8f;

union ABfrag { v16bf v; v8bf h[2]; };

__device__ __forceinline__ v8f zero_v8f() {
    v8f a;
#pragma unroll
    for (int j = 0; j < 8; ++j) a[j] = 0.0f;
    return a;
}

// 16B async copy global -> LDS (per lane), tracked by ASYNCcnt.
// lds_addr: low 32 bits of a generic pointer into __shared__ (== LDS offset).
__device__ __forceinline__ void async_copy16(unsigned lds_addr, const void* gptr) {
    asm volatile("global_load_async_to_lds_b128 %0, %1, off"
                 :: "v"(lds_addr), "v"(gptr)
                 : "memory");
}

// ---------------------------------------------------------------------------
// f32 -> bf16 straight conversion, 4 elements/thread (float4 in, 8B out)
// ---------------------------------------------------------------------------
__global__ __launch_bounds__(256) void conv_bf16_kernel(
    const float* __restrict__ in, __bf16* __restrict__ out)
{
    size_t i = (size_t)blockIdx.x * 256 + threadIdx.x;
    float4 v = ((const float4*)in)[i];
    v4bf o;
    o[0] = (__bf16)v.x; o[1] = (__bf16)v.y; o[2] = (__bf16)v.z; o[3] = (__bf16)v.w;
    ((v4bf*)out)[i] = o;
}

// ---------------------------------------------------------------------------
// f32 [Kd][N] row-major -> bf16 [N][Kd] (transpose+convert), 32x32 LDS tile
// ---------------------------------------------------------------------------
__global__ __launch_bounds__(256) void transpose_conv_kernel(
    const float* __restrict__ W, __bf16* __restrict__ Wt, int Kd, int N)
{
    __shared__ float s[32][33];
    const int t = threadIdx.x, tx = t & 31, ty = t >> 5;
    const int n0 = blockIdx.x * 32, k0 = blockIdx.y * 32;
#pragma unroll
    for (int i = 0; i < 4; ++i)
        s[ty + 8 * i][tx] = W[(size_t)(k0 + ty + 8 * i) * N + (n0 + tx)];
    __syncthreads();
#pragma unroll
    for (int i = 0; i < 4; ++i)
        Wt[(size_t)(n0 + ty + 8 * i) * Kd + (k0 + tx)] = (__bf16)s[tx][ty + 8 * i];
}

// ---------------------------------------------------------------------------
// codebook prep: bf16 copy + ||c_k||^2 (one block per code, 256 thr = D_Z)
// ---------------------------------------------------------------------------
__global__ __launch_bounds__(256) void cb_prep_kernel(
    const float* __restrict__ cb, __bf16* __restrict__ cbb, float* __restrict__ cnorm)
{
    const int k = blockIdx.x, t = threadIdx.x;
    float v = cb[(size_t)k * DZ + t];
    cbb[(size_t)k * DZ + t] = (__bf16)v;
    float s = v * v;
#pragma unroll
    for (int off = 16; off; off >>= 1) s += __shfl_down(s, off, 32);
    __shared__ float red[8];
    const int lane = t & 31, w = t >> 5;
    if (lane == 0) red[w] = s;
    __syncthreads();
    if (w == 0) {
        float r = (lane < 8) ? red[lane] : 0.0f;
#pragma unroll
        for (int off = 4; off; off >>= 1) r += __shfl_down(r, off, 32);
        if (lane == 0) cnorm[k] = r;
    }
}

// ---------------------------------------------------------------------------
// Tiled bf16 WMMA GEMM:  C[M,N] = act(A[M,Kd] * Bt[N,Kd]^T + bias)
// 256 thr (8 waves), 128x128 tile, BK=32, wave = 32x64 subtile (2x4 frags).
// Tiles staged with async global->LDS copies, double-buffered on ASYNCcnt:
// 4 async b128 per wave per chunk; s_wait_asynccnt 4 retires the current
// chunk while the next chunk's copies fly behind the 8 WMMAs.
// ACT: 0=none, 1=relu, 2=sigmoid.  OUT: float or __bf16.
// ---------------------------------------------------------------------------
template <int ACT, typename OUT>
__global__ __launch_bounds__(256) void gemm_bf16_wmma(
    const __bf16* __restrict__ A, const __bf16* __restrict__ Bt,
    const float* __restrict__ bias, OUT* __restrict__ C,
    int M, int N, int Kd)
{
    constexpr int BM = 128, BN = 128, BK = 32, LDL = BK + 8;
    __shared__ __bf16 As[2][BM][LDL];
    __shared__ __bf16 Bs[2][BN][LDL];

    const int tid  = threadIdx.x;
    const int lane = tid & 31, w = tid >> 5;
    const int wm = w & 3, wn = w >> 2;     // 4x2 wave grid over 128x128
    const int l16 = lane & 15, hi = lane >> 4;
    const int r0 = blockIdx.y * BM, c0 = blockIdx.x * BN;

    v8f acc[2][4];
#pragma unroll
    for (int mf = 0; mf < 2; ++mf)
#pragma unroll
        for (int nf = 0; nf < 4; ++nf) acc[mf][nf] = zero_v8f();

    // this lane's two 16B tile slots (s = slot index among 512 per matrix)
    const int s0  = (w * 2 + 0) * 32 + lane;
    const int s1  = (w * 2 + 1) * 32 + lane;
    const int row0 = s0 >> 2, ko0 = (s0 & 3) * 8;
    const int row1 = s1 >> 2, ko1 = (s1 & 3) * 8;

    const int nchunks = Kd / BK;

    // issue the 4 async tile copies (A x2, B x2) for chunk kc into buffer bb
    auto issue_tile = [&](int kc, int bb) {
        const int k0 = kc * BK;
        async_copy16((unsigned)(uintptr_t)&As[bb][row0][ko0],
                     A  + (size_t)(r0 + row0) * Kd + k0 + ko0);
        async_copy16((unsigned)(uintptr_t)&Bs[bb][row0][ko0],
                     Bt + (size_t)(c0 + row0) * Kd + k0 + ko0);
        async_copy16((unsigned)(uintptr_t)&As[bb][row1][ko1],
                     A  + (size_t)(r0 + row1) * Kd + k0 + ko1);
        async_copy16((unsigned)(uintptr_t)&Bs[bb][row1][ko1],
                     Bt + (size_t)(c0 + row1) * Kd + k0 + ko1);
    };

    issue_tile(0, 0);

    for (int kc = 0; kc < nchunks; ++kc) {
        const int cur = kc & 1;
        if (kc + 1 < nchunks) {
            issue_tile(kc + 1, cur ^ 1);
            // 4 outstanding = next chunk only => current chunk landed in LDS
            asm volatile("s_wait_asynccnt 0x4" ::: "memory");
        } else {
            asm volatile("s_wait_asynccnt 0x0" ::: "memory");
        }
        __syncthreads();

        // A frag (16x32 bf16): lanes 0-15 rows M, K 0-7 & 16-23; lanes 16-31 K 8-15 & 24-31
        ABfrag a[2], b[4];
#pragma unroll
        for (int mf = 0; mf < 2; ++mf) {
            const __bf16* p = &As[cur][32 * wm + 16 * mf + l16][hi * 8];
            a[mf].h[0] = *(const v8bf*)p;
            a[mf].h[1] = *(const v8bf*)(p + 16);
        }
        // B frag (32x16 bf16): lanes 0-15 col N, K 0-7 & 16-23; lanes 16-31 K 8-15 & 24-31
#pragma unroll
        for (int nf = 0; nf < 4; ++nf) {
            const __bf16* p = &Bs[cur][64 * wn + 16 * nf + l16][hi * 8];
            b[nf].h[0] = *(const v8bf*)p;
            b[nf].h[1] = *(const v8bf*)(p + 16);
        }
#pragma unroll
        for (int mf = 0; mf < 2; ++mf)
#pragma unroll
            for (int nf = 0; nf < 4; ++nf)
                acc[mf][nf] = __builtin_amdgcn_wmma_f32_16x16x32_bf16(
                    false, a[mf].v, false, b[nf].v, (short)0, acc[mf][nf], false, false);
        __syncthreads();  // protects WAR on the buffer refilled next iteration
    }

    // epilogue: bias + activation, C lane layout (row = d + 8*hi, col = l16)
#pragma unroll
    for (int nf = 0; nf < 4; ++nf) {
        const int col = c0 + 64 * wn + 16 * nf + l16;
        const float bv = bias[col];
#pragma unroll
        for (int mf = 0; mf < 2; ++mf) {
            const int rb = r0 + 32 * wm + 16 * mf + 8 * hi;
#pragma unroll
            for (int d = 0; d < 8; ++d) {
                float v = acc[mf][nf][d] + bv;
                if (ACT == 1) v = fmaxf(v, 0.0f);
                if (ACT == 2) v = 1.0f / (1.0f + __expf(-v));
                C[(size_t)(rb + d) * N + col] = (OUT)v;
            }
        }
    }
}

// ---------------------------------------------------------------------------
// VQ argmin: per 16-row tile, scores = z * cb^T via WMMA over K=8192 codes.
// argmin_k (||c_k||^2 - 2 z.c_k). Wave keeps z fragments resident (8x v16bf).
// ---------------------------------------------------------------------------
__global__ __launch_bounds__(256) void vq_argmin_kernel(
    const float* __restrict__ z, const __bf16* __restrict__ cbb,
    const float* __restrict__ cnorm, int* __restrict__ idx_out)
{
    const int lane = threadIdx.x & 31, w = threadIdx.x >> 5;
    const int l16 = lane & 15, hi = lane >> 4;
    const int r0 = blockIdx.x * 128 + w * 16;
    const int row = r0 + l16;

    // z fragments for K-dim 0..255 (8 chunks of 32), converted f32->bf16
    ABfrag zf[8];
    const int kbA = hi * 8;
#pragma unroll
    for (int c = 0; c < 8; ++c) {
        const float* zr = z + (size_t)row * DZ + 32 * c + kbA;
#pragma unroll
        for (int j = 0; j < 8; ++j) {
            zf[c].v[j]     = (__bf16)zr[j];
            zf[c].v[8 + j] = (__bf16)zr[16 + j];
        }
    }

    float bestv[8]; int besti[8];
#pragma unroll
    for (int d = 0; d < 8; ++d) { bestv[d] = 3.4e38f; besti[d] = 0; }

    const int kbB = hi * 8;
    for (int t = 0; t < KCB / 16; ++t) {
        const int code = t * 16 + l16;           // this lane's column/code
        v8f acc = zero_v8f();
#pragma unroll
        for (int c = 0; c < 8; ++c) {
            ABfrag bf;
            const v8bf* bp = (const v8bf*)(cbb + (size_t)code * DZ + 32 * c + kbB);
            bf.h[0] = bp[0];
            bf.h[1] = bp[2];
            acc = __builtin_amdgcn_wmma_f32_16x16x32_bf16(
                false, zf[c].v, false, bf.v, (short)0, acc, false, false);
        }
        const float cn = cnorm[code];
#pragma unroll
        for (int d = 0; d < 8; ++d) {
            float val = cn - 2.0f * acc[d];
            bool better = val < bestv[d];
            bestv[d] = better ? val : bestv[d];
            besti[d] = better ? code : besti[d];
        }
    }

    // min-reduce (val, idx) across the 16 lanes holding the same row
#pragma unroll
    for (int d = 0; d < 8; ++d) {
#pragma unroll
        for (int off = 1; off < 16; off <<= 1) {
            float ov = __shfl_xor(bestv[d], off, 32);
            int   oi = __shfl_xor(besti[d], off, 32);
            if (ov < bestv[d] || (ov == bestv[d] && oi < besti[d])) {
                bestv[d] = ov; besti[d] = oi;
            }
        }
        if (l16 == 0) idx_out[r0 + d + 8 * hi] = besti[d];
    }
}

// ---------------------------------------------------------------------------
// gather q (bf16), accumulate vq loss partials, emit idx as float
// one block per batch row; 256 thr = D_Z
// ---------------------------------------------------------------------------
__global__ __launch_bounds__(256) void gather_loss_kernel(
    const float* __restrict__ z, const float* __restrict__ cb,
    const int* __restrict__ idx, __bf16* __restrict__ qb,
    float* __restrict__ loss_acc, float* __restrict__ idx_out_f)
{
    const int b = blockIdx.x, t = threadIdx.x;
    const int k = idx[b];
    float c  = cb[(size_t)k * DZ + t];
    float zv = z[(size_t)b * DZ + t];
    qb[(size_t)b * DZ + t] = (__bf16)c;
    float d = c - zv;
    float s = d * d;
#pragma unroll
    for (int off = 16; off; off >>= 1) s += __shfl_down(s, off, 32);
    __shared__ float red[8];
    const int lane = t & 31, w = t >> 5;
    if (lane == 0) red[w] = s;
    __syncthreads();
    if (w == 0) {
        float r = (lane < 8) ? red[lane] : 0.0f;
#pragma unroll
        for (int off = 4; off; off >>= 1) r += __shfl_down(r, off, 32);
        if (lane == 0) atomicAdd(loss_acc, r);
    }
    if (t == 0) idx_out_f[b] = (float)k;
}

__global__ void loss_init_kernel(float* p) { *p = 0.0f; }

__global__ void loss_fin_kernel(const float* p, float* out)
{
    // vq_loss = q_loss + 0.25*e_loss = 1.25 * mean((q - z)^2)
    *out = (*p) * (1.25f / ((float)BATCH * (float)DZ));
}

// ---------------------------------------------------------------------------
extern "C" void kernel_launch(void* const* d_in, const int* in_sizes, int n_in,
                              void* d_out, int out_size, void* d_ws, size_t ws_size,
                              hipStream_t stream) {
    (void)in_sizes; (void)n_in; (void)out_size; (void)ws_size;
    const float* x  = (const float*)d_in[0];
    const float* W1 = (const float*)d_in[1];
    const float* b1 = (const float*)d_in[2];
    const float* W2 = (const float*)d_in[3];
    const float* b2 = (const float*)d_in[4];
    const float* cb = (const float*)d_in[5];
    const float* W3 = (const float*)d_in[6];
    const float* b3 = (const float*)d_in[7];
    const float* W4 = (const float*)d_in[8];
    const float* b4 = (const float*)d_in[9];
    float* out = (float*)d_out;

    char* ws = (char*)d_ws;
    size_t off = 0;
    auto alloc = [&](size_t bytes) -> void* {
        void* p = ws + off;
        off = (off + bytes + 255) & ~(size_t)255;
        return p;
    };
    __bf16* xb    = (__bf16*)alloc((size_t)BATCH * DIN * 2);   // 128 MB (L2-resident)
    __bf16* w1t   = (__bf16*)alloc((size_t)DH * DIN * 2);      // W1^T [DH][DIN]
    __bf16* w2t   = (__bf16*)alloc((size_t)DZ * DH * 2);       // W2^T [DZ][DH]
    __bf16* w3t   = (__bf16*)alloc((size_t)DH * DZ * 2);       // W3^T [DH][DZ]
    __bf16* w4t   = (__bf16*)alloc((size_t)DIN * DH * 2);      // W4^T [DIN][DH]
    __bf16* cbb   = (__bf16*)alloc((size_t)KCB * DZ * 2);      // codebook bf16
    float*  cnorm = (float*) alloc((size_t)KCB * 4);
    __bf16* h1    = (__bf16*)alloc((size_t)BATCH * DH * 2);    // enc hidden (also reused as h2)
    float*  zf    = (float*) alloc((size_t)BATCH * DZ * 4);
    int*    idxw  = (int*)   alloc((size_t)BATCH * 4);
    __bf16* qb    = (__bf16*)alloc((size_t)BATCH * DZ * 2);
    float*  lacc  = (float*) alloc(4);
    __bf16* h2    = h1;  // h1 is dead after GEMM2; alias to save workspace

    float* out_loss = out + (size_t)BATCH * DIN;
    float* out_idx  = out_loss + 1;

    // 1) precision conversion / weight transposes (one-time, bandwidth-trivial)
    conv_bf16_kernel<<<dim3((BATCH * (size_t)DIN) / 4 / 256), 256, 0, stream>>>(x, xb);
    transpose_conv_kernel<<<dim3(DH / 32,  DIN / 32), 256, 0, stream>>>(W1, w1t, DIN, DH);
    transpose_conv_kernel<<<dim3(DZ / 32,  DH  / 32), 256, 0, stream>>>(W2, w2t, DH,  DZ);
    transpose_conv_kernel<<<dim3(DH / 32,  DZ  / 32), 256, 0, stream>>>(W3, w3t, DZ,  DH);
    transpose_conv_kernel<<<dim3(DIN / 32, DH  / 32), 256, 0, stream>>>(W4, w4t, DH,  DIN);
    cb_prep_kernel<<<dim3(KCB), 256, 0, stream>>>(cb, cbb, cnorm);

    // 2) encoder: h1 = relu(x@W1+b1), z = h1@W2+b2
    gemm_bf16_wmma<1, __bf16><<<dim3(DH / 128, BATCH / 128), 256, 0, stream>>>(
        xb, w1t, b1, h1, BATCH, DH, DIN);
    gemm_bf16_wmma<0, float><<<dim3(DZ / 128, BATCH / 128), 256, 0, stream>>>(
        h1, w2t, b2, zf, BATCH, DZ, DH);

    // 3) VQ: WMMA score sweep + fused argmin
    vq_argmin_kernel<<<dim3(BATCH / 128), 256, 0, stream>>>(zf, cbb, cnorm, idxw);

    // 4) gather codes, vq loss, idx output
    loss_init_kernel<<<dim3(1), 1, 0, stream>>>(lacc);
    gather_loss_kernel<<<dim3(BATCH), 256, 0, stream>>>(zf, cb, idxw, qb, lacc, out_idx);

    // 5) decoder: h2 = relu(q@W3+b3), x_recon = sigmoid(h2@W4+b4)
    gemm_bf16_wmma<1, __bf16><<<dim3(DH / 128, BATCH / 128), 256, 0, stream>>>(
        qb, w3t, b3, h2, BATCH, DH, DZ);
    gemm_bf16_wmma<2, float><<<dim3(DIN / 128, BATCH / 128), 256, 0, stream>>>(
        h2, w4t, b4, out, BATCH, DIN, DH);

    loss_fin_kernel<<<dim3(1), 1, 0, stream>>>(lacc, out_loss);
}
